// Self_Attention_37735582662921
// MI455X (gfx1250) — matrix operand order repaired and checked by reading, compiled, not run on hardware
//
#include <hip/hip_runtime.h>

typedef __bf16 bf16;
typedef __attribute__((ext_vector_type(16))) __bf16 v16bf;
typedef __attribute__((ext_vector_type(8)))  __bf16 v8bf;
typedef __attribute__((ext_vector_type(4)))  __bf16 v4bf;
typedef __attribute__((ext_vector_type(8)))  float  v8f;
typedef __attribute__((ext_vector_type(4)))  float  v4f;
typedef __attribute__((ext_vector_type(4)))  int    v4i;

#define NB 8      // batches
#define SS 2048   // sequence
#define SD 768    // model dim
#define SO 768    // output dim

enum { OUT_F32 = 0, OUT_BF16 = 1, OUT_BF16T = 2 };

#if __has_builtin(__builtin_amdgcn_global_load_async_to_lds_b128)
#define HAVE_ASYNC 1
#else
#define HAVE_ASYNC 0
#endif

__device__ __forceinline__ void stage_b128(const bf16* g, bf16* l)
{
#if HAVE_ASYNC
    __builtin_amdgcn_global_load_async_to_lds_b128(
        (__attribute__((address_space(1))) v4i*)(g),
        (__attribute__((address_space(3))) v4i*)(l), 0, 0);
#else
    *(v8bf*)l = *(const v8bf*)g;
#endif
}

template<int N>
__device__ __forceinline__ void wait_async()
{
#if HAVE_ASYNC
#if __has_builtin(__builtin_amdgcn_s_wait_asynccnt)
    __builtin_amdgcn_s_wait_asynccnt(N);
#else
    asm volatile("s_wait_asynccnt %0" :: "i"(N) : "memory");
#endif
#endif
}

// ---------------------------------------------------------------------------
// Tiled bf16 WMMA GEMM:  C[z][M][N] = scale * A[z][M][K] * B[z][K][N]
// B supplied TRANSPOSED (Bt, row-major [N][K]).
// Block tile 128x256x32, 256 threads = 8 waves (2 M x 4 N), wave tile 64x64.
// Double-buffered LDS filled with GLOBAL_LOAD_ASYNC_TO_LDS_B128.
// ---------------------------------------------------------------------------
template<int OUT_MODE>
__global__ __launch_bounds__(256)
void gemm_wmma_bf16(const bf16* __restrict__ A, long lda, long strideA,
                    const bf16* __restrict__ Bt, long ldb, long strideB,
                    void*       __restrict__ Cv, long ldc, long strideC,
                    int K, float scale)
{
    const int tid  = threadIdx.x;
    const int lane = tid & 31;
    const int wave = tid >> 5;
    const int l16  = lane & 15;
    const int half = lane >> 4;

    const long z  = blockIdx.z;
    const int  bm = blockIdx.y * 128;
    const int  bn = blockIdx.x * 256;

    const int wm = (wave >> 2) * 64;
    const int wn = (wave & 3) * 64;

    const bf16* Ap = A  + z * strideA;
    const bf16* Bp = Bt + z * strideB;

    // pitch 40 elems (80B): conflict-free b128 reads across 16 consecutive rows
    __shared__ __align__(16) bf16 lA[2][128 * 40];
    __shared__ __align__(16) bf16 lB[2][256 * 40];

    v8f acc[4][4] = {};

    const int nIter = K >> 5;

    // ---- stage one 128x32 A slab + 256x32 B slab: 6 b128 per thread ----
    auto stage = [&](int buf, int kt) {
        #pragma unroll
        for (int it = 0; it < 2; ++it) {               // A: 128 rows * 64B
            int g  = tid + it * 256;
            int m  = g >> 2;
            int kg = (g & 3) << 3;
            stage_b128(Ap + (long)(bm + m) * lda + kt + kg, &lA[buf][m * 40 + kg]);
        }
        #pragma unroll
        for (int it = 0; it < 4; ++it) {               // B: 256 rows * 64B
            int g  = tid + it * 256;
            int n  = g >> 2;
            int kg = (g & 3) << 3;
            stage_b128(Bp + (long)(bn + n) * ldb + kt + kg, &lB[buf][n * 40 + kg]);
        }
    };

    stage(0, 0);

    for (int it = 0; it < nIter; ++it) {
        const int cur = it & 1;
        const bool has_next = (it + 1) < nIter;
        if (has_next) {
            stage(1 - cur, (it + 1) << 5);
            wait_async<6>();        // previous tile landed; 6 new in flight
        } else {
            wait_async<0>();
        }
        __syncthreads();            // slab visible to whole workgroup

        // A 16x32 frag: lane -> row l16, K runs [h*8,+8) and [16+h*8,+8)
        v16bf a[4];
        #pragma unroll
        for (int i = 0; i < 4; ++i) {
            const bf16* p = &lA[cur][(wm + i * 16 + l16) * 40];
            v8bf lo = *(const v8bf*)(p + half * 8);
            v8bf hi = *(const v8bf*)(p + 16 + half * 8);
            a[i] = __builtin_shufflevector(lo, hi, 0,1,2,3,4,5,6,7,
                                                   8,9,10,11,12,13,14,15);
        }
        // B 32x16 frag: lane -> col l16, K runs [h*8,+8) and [16+h*8,+8)
        v16bf b[4];
        #pragma unroll
        for (int j = 0; j < 4; ++j) {
            const bf16* p = &lB[cur][(wn + j * 16 + l16) * 40];
            v8bf lo = *(const v8bf*)(p + half * 8);
            v8bf hi = *(const v8bf*)(p + 16 + half * 8);
            b[j] = __builtin_shufflevector(lo, hi, 0,1,2,3,4,5,6,7,
                                                   8,9,10,11,12,13,14,15);
        }

        #pragma unroll
        for (int i = 0; i < 4; ++i)
            #pragma unroll
            for (int j = 0; j < 4; ++j)
                acc[i][j] = __builtin_amdgcn_wmma_f32_16x16x32_bf16(
                    false, a[i], false, b[j], (short)0, acc[i][j], false, false);

        __syncthreads();            // everyone done reading before overwrite
    }

    // ---- epilogue: C frag: VGPR r -> M = r + 8*half, N = l16 ----
    #pragma unroll
    for (int i = 0; i < 4; ++i) {
        #pragma unroll
        for (int j = 0; j < 4; ++j) {
            int m0 = bm + wm + i * 16 + half * 8;
            int n  = bn + wn + j * 16 + l16;
            if (OUT_MODE == OUT_F32) {
                float* Cp = (float*)Cv + z * strideC;
                #pragma unroll
                for (int r = 0; r < 8; ++r)
                    Cp[(long)(m0 + r) * ldc + n] = acc[i][j][r] * scale;
            } else if (OUT_MODE == OUT_BF16) {
                bf16* Cp = (bf16*)Cv + z * strideC;
                #pragma unroll
                for (int r = 0; r < 8; ++r)
                    Cp[(long)(m0 + r) * ldc + n] = (bf16)(acc[i][j][r] * scale);
            } else { // OUT_BF16T: C is [N][M]; 8 consecutive M -> one 16B store
                bf16* Cp = (bf16*)Cv + z * strideC;
                v8bf pk;
                #pragma unroll
                for (int r = 0; r < 8; ++r)
                    pk[r] = (bf16)(acc[i][j][r] * scale);
                *(v8bf*)(Cp + (long)n * ldc + m0) = pk;
            }
        }
    }
}

// fp32 x -> bf16 (flat, 4 elems/thread)
__global__ __launch_bounds__(256)
void xconv_kernel(const float* __restrict__ x, bf16* __restrict__ xb)
{
    long g = (long)blockIdx.x * 256 + threadIdx.x;
    v4f v = *(const v4f*)(x + g * 4);
    v4bf c;
    c[0] = (bf16)v[0]; c[1] = (bf16)v[1]; c[2] = (bf16)v[2]; c[3] = (bf16)v[3];
    *(v4bf*)(xb + g * 4) = c;
}

// fp32 kernel[3][D][O] -> bf16 Wt[3][O][D] (transposed per matrix)
__global__ __launch_bounds__(256)
void wconv_kernel(const float* __restrict__ W, bf16* __restrict__ Wt)
{
    long idx = (long)blockIdx.x * 256 + threadIdx.x;
    int  m = (int)(idx / (SD * SO));
    int  r = (int)(idx % (SD * SO));
    int  d = r / SO;
    int  o = r % SO;
    Wt[(long)m * SO * SD + (long)o * SD + d] = (bf16)W[idx];
}

// One block per score row: softmax(2048 f32) -> bf16 written in-place at row
// start (row pitch stays 2048 f32 = 4096 bf16 for the AV GEMM).
__global__ __launch_bounds__(256)
void softmax_kernel(float* __restrict__ scores)
{
    const int tid = threadIdx.x;
    float* row = scores + (long)blockIdx.x * SS;

    float v[8];
    float mx = -3.402823e38f;
    #pragma unroll
    for (int e = 0; e < 8; ++e) {
        v[e] = row[tid + e * 256];
        mx = fmaxf(mx, v[e]);
    }

    __shared__ float red[256];
    red[tid] = mx;
    __syncthreads();
    for (int s = 128; s > 0; s >>= 1) {
        if (tid < s) red[tid] = fmaxf(red[tid], red[tid + s]);
        __syncthreads();
    }
    float m = red[0];
    __syncthreads();

    float sum = 0.f;
    #pragma unroll
    for (int e = 0; e < 8; ++e) {
        v[e] = expf(v[e] - m);
        sum += v[e];
    }
    red[tid] = sum;
    __syncthreads();
    for (int s = 128; s > 0; s >>= 1) {
        if (tid < s) red[tid] += red[tid + s];
        __syncthreads();
    }
    float inv = 1.0f / red[0];

    bf16* out = (bf16*)row;               // in-place; all reads already done
    #pragma unroll
    for (int e = 0; e < 8; ++e)
        out[tid + e * 256] = (bf16)(v[e] * inv);
}

extern "C" void kernel_launch(void* const* d_in, const int* in_sizes, int n_in,
                              void* d_out, int out_size, void* d_ws, size_t ws_size,
                              hipStream_t stream)
{
    const float* x = (const float*)d_in[0];   // [8,2048,768]
    const float* W = (const float*)d_in[1];   // [3,768,768]
    float* out = (float*)d_out;               // [8,2048,768]

    char* ws = (char*)d_ws;
    float* scores = (float*)(ws);                       // 128 MiB
    bf16*  Qb     = (bf16*)(ws + 134217728L);           // 24 MiB
    bf16*  Kb     = (bf16*)(ws + 159383552L);           // 24 MiB
    bf16*  Vt     = (bf16*)(ws + 184549376L);           // [b][O][S], 24 MiB
    bf16*  Wt     = (bf16*)(ws + 209715200L);           // 3*768*768 bf16
    bf16*  xb     = (bf16*)(ws + 213254144L);           // 24 MiB

    dim3 blk(256);

    // 0) input conversions
    xconv_kernel<<<(NB * SS * SD) / (256 * 4), blk, 0, stream>>>(x, xb);
    wconv_kernel<<<(3 * SD * SO) / 256, blk, 0, stream>>>(W, Wt);

    // 1) projections: Q, K normal bf16; V transposed bf16
    dim3 gproj(SO / 256, SS / 128, NB);
    gemm_wmma_bf16<OUT_BF16><<<gproj, blk, 0, stream>>>(
        xb, SD, (long)SS * SD, Wt + 0L * SD * SO, SD, 0,
        Qb, SO, (long)SS * SO, SD, 1.0f);
    gemm_wmma_bf16<OUT_BF16><<<gproj, blk, 0, stream>>>(
        xb, SD, (long)SS * SD, Wt + 1L * SD * SO, SD, 0,
        Kb, SO, (long)SS * SO, SD, 1.0f);
    gemm_wmma_bf16<OUT_BF16T><<<gproj, blk, 0, stream>>>(
        xb, SD, (long)SS * SD, Wt + 2L * SD * SO, SD, 0,
        Vt, SS, (long)SO * SS, SD, 1.0f);

    // 2) scores = (Q @ K^T) * 1/sqrt(64); K rows ARE the K^T columns
    dim3 gqk(SS / 256, SS / 128, NB);
    gemm_wmma_bf16<OUT_F32><<<gqk, blk, 0, stream>>>(
        Qb, SO, (long)SS * SO, Kb, SO, (long)SS * SO,
        scores, SS, (long)SS * SS, SO, 0.125f);

    // 3) row softmax -> bf16 P in-place (row pitch 4096 bf16)
    softmax_kernel<<<NB * SS, blk, 0, stream>>>(scores);

    // 4) out = P @ V  (Vt rows are the V columns)
    dim3 gav(SO / 256, SS / 128, NB);
    gemm_wmma_bf16<OUT_F32><<<gav, blk, 0, stream>>>(
        (const bf16*)scores, 2L * SS, 2L * (long)SS * SS,
        Vt, SS, (long)SO * SS,
        out, SO, (long)SS * SO, SS, 1.0f);
}
